// Token_dend_QK_Attention_12953621364901
// MI455X (gfx1250) — matrix operand order repaired and checked by reading, not compiled, not run on hardware
//
#include <hip/hip_runtime.h>
#include <hip/hip_bf16.h>

typedef __attribute__((ext_vector_type(16))) __bf16 v16bf;
typedef __attribute__((ext_vector_type(8)))  __bf16 v8bf;
typedef __attribute__((ext_vector_type(4)))  __bf16 v4bf;
typedef __attribute__((ext_vector_type(8)))  float  v8f;

#define T_     4
#define B_     32
#define C_     384
#define HW_    196
#define HWP_   224      // HW padded to a multiple of 32 (pad rows are dont-care)
#define N_     128      // T_*B_
#define HEADS_ 8
#define DH_    48       // C_/HEADS_

// ---------------------------------------------------------------------------
// f32 -> bf16 conversion (vectorized by 4) — used for the three weight mats.
// ---------------------------------------------------------------------------
__global__ __launch_bounds__(256)
void cvt_f32_to_bf16(const float* __restrict__ in, __bf16* __restrict__ out, int n4) {
    int i = blockIdx.x * 256 + threadIdx.x;
    if (i < n4) {
        float4 v = ((const float4*)in)[i];
        v4bf o;
        o[0] = (__bf16)v.x; o[1] = (__bf16)v.y; o[2] = (__bf16)v.z; o[3] = (__bf16)v.w;
        ((v4bf*)out)[i] = o;
    }
}

// ---------------------------------------------------------------------------
// x (N,C,HW) f32  ->  xt (N,HWP,C) bf16   (transpose + convert, LDS tiled)
// Pad rows [196,224) are left untouched (never stored from, see GEMMs).
// Grid: (HWP/32=7, C/32=12, N=128), block 256.
// ---------------------------------------------------------------------------
__global__ __launch_bounds__(256)
void transpose_cvt(const float* __restrict__ x, __bf16* __restrict__ xt) {
    __shared__ float tile[32][33];
    const int tid = threadIdx.x;
    const int h0 = blockIdx.x * 32;
    const int c0 = blockIdx.y * 32;
    const int n  = blockIdx.z;

    // load 32(c) x 32(hw), hw contiguous; 196 % 4 == 0 so float4 chunks are
    // all-in or all-out of range.
    {
        int r    = tid >> 3;          // c offset 0..31
        int col0 = (tid & 7) * 4;     // hw offset 0,4,...,28
        float4 v = make_float4(0.f, 0.f, 0.f, 0.f);
        if (h0 + col0 < HW_)
            v = *(const float4*)&x[((size_t)n * C_ + c0 + r) * HW_ + h0 + col0];
        tile[r][col0 + 0] = v.x; tile[r][col0 + 1] = v.y;
        tile[r][col0 + 2] = v.z; tile[r][col0 + 3] = v.w;
    }
    __syncthreads();
    // store transposed: 32(hw) x 32(c), c contiguous, as bf16
    {
        int r2 = tid >> 3;            // hw offset 0..31
        int c2 = (tid & 7) * 4;       // c offset
        int ghw = h0 + r2;
        if (ghw < HW_) {
            v4bf o;
            o[0] = (__bf16)tile[c2 + 0][r2];
            o[1] = (__bf16)tile[c2 + 1][r2];
            o[2] = (__bf16)tile[c2 + 2][r2];
            o[3] = (__bf16)tile[c2 + 3][r2];
            *(v4bf*)&xt[((size_t)n * HWP_ + ghw) * C_ + c0 + c2] = o;
        }
    }
}

// ---------------------------------------------------------------------------
// Fused Q+K 1x1-conv GEMM + BN.  B tile (x) is loaded once, used twice.
//   xt : (N, HWP, C) bf16 ; wq/wk : (C_, C_) bf16 ; q/k out: (N, C, HW) bf16
// Block 256 = 8 waves, tile 64(O) x 32(HW), K by 32, double-buffered LDS,
// register-staged pipeline, one barrier per K-step.  No bounds checks in the
// hot loop: pad rows feed only never-stored output columns.
// ---------------------------------------------------------------------------
__global__ __launch_bounds__(256)
void gemm_qk(const __bf16* __restrict__ xt,
             const __bf16* __restrict__ wq, const __bf16* __restrict__ wk,
             const float* __restrict__ qg, const float* __restrict__ qbe,
             const float* __restrict__ qm, const float* __restrict__ qv,
             const float* __restrict__ kg, const float* __restrict__ kbe,
             const float* __restrict__ km, const float* __restrict__ kv,
             __bf16* __restrict__ qout, __bf16* __restrict__ kout) {
    __shared__ __align__(16) __bf16 lAq[2][64][40];
    __shared__ __align__(16) __bf16 lAk[2][64][40];
    __shared__ __align__(16) __bf16 lB [2][32][40];
    __shared__ float sScQ[64], sShQ[64], sScK[64], sShK[64];

    const int tid = threadIdx.x;
    const int hw0 = blockIdx.x * 32;
    const int o0  = blockIdx.y * 64;
    const int n   = blockIdx.z;

    if (tid < 64) {
        int o = o0 + tid;
        float ivq = qg[o] * rsqrtf(qv[o] + 1e-5f);
        sScQ[tid] = ivq; sShQ[tid] = qbe[o] - qm[o] * ivq;
        float ivk = kg[o] * rsqrtf(kv[o] + 1e-5f);
        sScK[tid] = ivk; sShK[tid] = kbe[o] - km[o] * ivk;
    }

    const int wv = tid >> 5, lane = tid & 31;
    const int wm = wv & 3, wn = wv >> 2;

    // staging indices
    const int ar = tid >> 2;            // 0..63 (O row)
    const int ak = (tid & 3) * 8;       // k offset
    const int br = tid >> 3;            // 0..31 (hw row)
    const int bk = (tid & 7) * 4;       // k offset

    const __bf16* pAq = &wq[(size_t)(o0 + ar) * C_ + ak];
    const __bf16* pAk = &wk[(size_t)(o0 + ar) * C_ + ak];
    const __bf16* pB  = &xt[((size_t)n * HWP_ + hw0 + br) * C_ + bk];

    v8bf rq = *(const v8bf*)pAq;
    v8bf rk = *(const v8bf*)pAk;
    v4bf rb = *(const v4bf*)pB;

    *(v8bf*)&lAq[0][ar][ak] = rq;
    *(v8bf*)&lAk[0][ar][ak] = rk;
    *(v4bf*)&lB [0][br][bk] = rb;
    __syncthreads();

    // fragment indices
    const int m   = wm * 16 + (lane & 15);
    const int kbA = (lane < 16) ? 0 : 8;
    const int nn  = wn * 16 + (lane & 15);
    const int kbB = (lane < 16) ? 0 : 8;

    v8f accq = {}, acck = {};
    int cur = 0;
    for (int kk = 32; kk <= C_; kk += 32) {
        const bool more = (kk < C_);
        if (more) {
            rq = *(const v8bf*)(pAq + kk);
            rk = *(const v8bf*)(pAk + kk);
            rb = *(const v4bf*)(pB + kk);
        }
        v8bf qlo = *(const v8bf*)&lAq[cur][m][kbA];
        v8bf qhi = *(const v8bf*)&lAq[cur][m][16 + kbA];
        v8bf klo = *(const v8bf*)&lAk[cur][m][kbA];
        v8bf khi = *(const v8bf*)&lAk[cur][m][16 + kbA];
        v8bf blo = *(const v8bf*)&lB [cur][nn][kbB];
        v8bf bhi = *(const v8bf*)&lB [cur][nn][kbB + 16];
        v16bf aq, akf, bf;
        #pragma unroll
        for (int e = 0; e < 8; ++e) {
            aq [e] = qlo[e]; aq [e + 8] = qhi[e];
            akf[e] = klo[e]; akf[e + 8] = khi[e];
            bf [e] = blo[e]; bf [e + 8] = bhi[e];
        }
        accq = __builtin_amdgcn_wmma_f32_16x16x32_bf16(false, aq,  false, bf, (short)0, accq, false, false);
        acck = __builtin_amdgcn_wmma_f32_16x16x32_bf16(false, akf, false, bf, (short)0, acck, false, false);
        if (more) {
            *(v8bf*)&lAq[cur ^ 1][ar][ak] = rq;
            *(v8bf*)&lAk[cur ^ 1][ar][ak] = rk;
            *(v4bf*)&lB [cur ^ 1][br][bk] = rb;
        }
        __syncthreads();
        cur ^= 1;
    }

    const int ncol = hw0 + wn * 16 + (lane & 15);
    if (ncol < HW_) {
        #pragma unroll
        for (int r = 0; r < 8; ++r) {
            int mloc = wm * 16 + ((lane < 16) ? r : (8 + r));
            size_t oidx = ((size_t)n * C_ + (o0 + mloc)) * HW_ + ncol;
            qout[oidx] = (__bf16)(accq[r] * sScQ[mloc] + sShQ[mloc]);
            kout[oidx] = (__bf16)(acck[r] * sScK[mloc] + sShK[mloc]);
        }
    }
}

// ---------------------------------------------------------------------------
// q-LIF -> head-sum -> attn-LIF(0.5) -> k-LIF -> y = attn*k, written as bf16
// in TRANSPOSED padded layout (N, HWP, C) for the projection GEMM.
// One block per (b, head); spikes staged as bytes in LDS for coalesced output.
// ---------------------------------------------------------------------------
__global__ __launch_bounds__(256)
void lif_attn(const __bf16* __restrict__ qpre, const __bf16* __restrict__ kpre,
              __bf16* __restrict__ yt) {
    __shared__ unsigned char spk[T_][DH_][HW_ + 4];   // 38.4 KB
    const int tid  = threadIdx.x;
    const int head = blockIdx.x % HEADS_;
    const int b    = blockIdx.x / HEADS_;
    const size_t tstride = (size_t)B_ * C_ * HW_;

    if (tid < HW_) {
        const int hw = tid;
        const size_t base = ((size_t)b * C_ + head * DH_) * HW_ + hw;
        float qs[T_] = {0.f, 0.f, 0.f, 0.f};
        for (int d = 0; d < DH_; ++d) {
            size_t p = base + (size_t)d * HW_;
            float v = 0.f;
            #pragma unroll
            for (int t = 0; t < T_; ++t) {
                float h = 0.5f * (v + (float)qpre[p + t * tstride]);
                float s = (h >= 1.0f) ? 1.f : 0.f;
                v = h * (1.f - s);
                qs[t] += s;
            }
        }
        float at[T_];
        {
            float v = 0.f;
            #pragma unroll
            for (int t = 0; t < T_; ++t) {
                float h = 0.5f * (v + qs[t]);
                float s = (h >= 0.5f) ? 1.f : 0.f;
                v = h * (1.f - s);
                at[t] = s;
            }
        }
        for (int d = 0; d < DH_; ++d) {
            size_t p = base + (size_t)d * HW_;
            float v = 0.f;
            #pragma unroll
            for (int t = 0; t < T_; ++t) {
                float h = 0.5f * (v + (float)kpre[p + t * tstride]);
                float s = (h >= 1.0f) ? 1.f : 0.f;
                v = h * (1.f - s);
                spk[t][d][hw] = (unsigned char)(s * at[t]);
            }
        }
    }
    __syncthreads();

    // emit yt[(t*B+b, hw, head*DH+d)] with (hw,d)-flattened coalesced stores
    const int tot = DH_ * HW_;   // 9408
    unsigned short* yo = (unsigned short*)yt;
    #pragma unroll
    for (int t = 0; t < T_; ++t) {
        size_t obase = ((size_t)(t * B_ + b) * HWP_) * C_ + head * DH_;
        for (int i = tid; i < tot; i += 256) {
            int hw = i / DH_, d = i % DH_;
            unsigned short v = spk[t][d][hw] ? (unsigned short)0x3F80 : (unsigned short)0;
            yo[obase + (size_t)hw * C_ + d] = v;
        }
    }
}

// ---------------------------------------------------------------------------
// Projection GEMM (+bias) + BN + final LIF fused.  Grid z = b (32); the 4
// timesteps share the A (weight) fragment -> 4 WMMAs per K-step, LIF scan in
// the epilogue writes d_out directly.
// ---------------------------------------------------------------------------
__global__ __launch_bounds__(256)
void gemm_proj_lif(const __bf16* __restrict__ yt,   // (N, HWP, C)
                   const __bf16* __restrict__ wp,
                   const float* __restrict__ pg, const float* __restrict__ pbe,
                   const float* __restrict__ pm, const float* __restrict__ pv,
                   const float* __restrict__ pb,
                   float* __restrict__ out) {       // (T,B,C,HW)
    __shared__ __align__(16) __bf16 lA[2][64][40];
    __shared__ __align__(16) __bf16 lB[2][T_][32][40];
    __shared__ float sSc[64], sSh[64];

    const int tid = threadIdx.x;
    const int hw0 = blockIdx.x * 32;
    const int o0  = blockIdx.y * 64;
    const int b   = blockIdx.z;

    if (tid < 64) {
        int o = o0 + tid;
        float iv = pg[o] * rsqrtf(pv[o] + 1e-5f);
        sSc[tid] = iv;
        sSh[tid] = pbe[o] - pm[o] * iv + pb[o] * iv;
    }

    const int wv = tid >> 5, lane = tid & 31;
    const int wm = wv & 3, wn = wv >> 2;

    const int ar = tid >> 2, ak = (tid & 3) * 8;
    const int br = tid >> 3, bk = (tid & 7) * 4;
    const size_t TS = (size_t)B_ * HWP_ * C_;   // timestep stride in yt

    const __bf16* pA = &wp[(size_t)(o0 + ar) * C_ + ak];
    const __bf16* pB = &yt[((size_t)b * HWP_ + hw0 + br) * C_ + bk];

    v8bf ra = *(const v8bf*)pA;
    v4bf rbt[T_];
    #pragma unroll
    for (int t = 0; t < T_; ++t) rbt[t] = *(const v4bf*)(pB + t * TS);

    *(v8bf*)&lA[0][ar][ak] = ra;
    #pragma unroll
    for (int t = 0; t < T_; ++t) *(v4bf*)&lB[0][t][br][bk] = rbt[t];
    __syncthreads();

    const int m   = wm * 16 + (lane & 15);
    const int kbA = (lane < 16) ? 0 : 8;
    const int nn  = wn * 16 + (lane & 15);
    const int kbB = (lane < 16) ? 0 : 8;

    v8f acc[T_] = {};
    int cur = 0;
    for (int kk = 32; kk <= C_; kk += 32) {
        const bool more = (kk < C_);
        if (more) {
            ra = *(const v8bf*)(pA + kk);
            #pragma unroll
            for (int t = 0; t < T_; ++t) rbt[t] = *(const v4bf*)(pB + kk + t * TS);
        }
        v8bf alo = *(const v8bf*)&lA[cur][m][kbA];
        v8bf ahi = *(const v8bf*)&lA[cur][m][16 + kbA];
        v16bf a;
        #pragma unroll
        for (int e = 0; e < 8; ++e) { a[e] = alo[e]; a[e + 8] = ahi[e]; }
        #pragma unroll
        for (int t = 0; t < T_; ++t) {
            v8bf blo = *(const v8bf*)&lB[cur][t][nn][kbB];
            v8bf bhi = *(const v8bf*)&lB[cur][t][nn][kbB + 16];
            v16bf bb;
            #pragma unroll
            for (int e = 0; e < 8; ++e) { bb[e] = blo[e]; bb[e + 8] = bhi[e]; }
            acc[t] = __builtin_amdgcn_wmma_f32_16x16x32_bf16(
                false, a, false, bb, (short)0, acc[t], false, false);
        }
        if (more) {
            *(v8bf*)&lA[cur ^ 1][ar][ak] = ra;
            #pragma unroll
            for (int t = 0; t < T_; ++t) *(v4bf*)&lB[cur ^ 1][t][br][bk] = rbt[t];
        }
        __syncthreads();
        cur ^= 1;
    }

    const int ncol = hw0 + wn * 16 + (lane & 15);
    if (ncol < HW_) {
        const size_t OTS = (size_t)B_ * C_ * HW_;   // timestep stride in out
        #pragma unroll
        for (int r = 0; r < 8; ++r) {
            int mloc = wm * 16 + ((lane < 16) ? r : (8 + r));
            float sc = sSc[mloc], sh = sSh[mloc];
            size_t obase = ((size_t)b * C_ + (o0 + mloc)) * HW_ + ncol;
            float v = 0.f;
            #pragma unroll
            for (int t = 0; t < T_; ++t) {
                float pre = acc[t][r] * sc + sh;
                float h = 0.5f * (v + pre);
                float s = (h >= 1.0f) ? 1.f : 0.f;
                v = h * (1.f - s);
                out[obase + t * OTS] = s;
            }
        }
    }
}

// ---------------------------------------------------------------------------
extern "C" void kernel_launch(void* const* d_in, const int* in_sizes, int n_in,
                              void* d_out, int out_size, void* d_ws, size_t ws_size,
                              hipStream_t stream) {
    const float* x   = (const float*)d_in[0];
    const float* qw  = (const float*)d_in[1];
    const float* qg  = (const float*)d_in[2];
    const float* qbe = (const float*)d_in[3];
    const float* qm  = (const float*)d_in[4];
    const float* qv  = (const float*)d_in[5];
    const float* kw  = (const float*)d_in[6];
    const float* kg  = (const float*)d_in[7];
    const float* kbe = (const float*)d_in[8];
    const float* km  = (const float*)d_in[9];
    const float* kv  = (const float*)d_in[10];
    const float* pw  = (const float*)d_in[11];
    const float* pb  = (const float*)d_in[12];
    const float* pg  = (const float*)d_in[13];
    const float* pbe = (const float*)d_in[14];
    const float* pm  = (const float*)d_in[15];
    const float* pv  = (const float*)d_in[16];
    (void)in_sizes; (void)n_in; (void)out_size; (void)ws_size;

    char* ws = (char*)d_ws;
    const size_t nElems  = (size_t)T_ * B_ * C_ * HW_;    // 9,633,792
    const size_t nElemsP = (size_t)N_ * HWP_ * C_;        // padded transposed

    __bf16* xt  = (__bf16*)ws;                            // (N,HWP,C); later yt
    size_t off  = nElemsP * sizeof(__bf16);
    __bf16* wqb = (__bf16*)(ws + off); off += (size_t)C_ * C_ * sizeof(__bf16);
    __bf16* wkb = (__bf16*)(ws + off); off += (size_t)C_ * C_ * sizeof(__bf16);
    __bf16* wpb = (__bf16*)(ws + off); off += (size_t)C_ * C_ * sizeof(__bf16);
    __bf16* qpre = (__bf16*)(ws + off); off += nElems * sizeof(__bf16);
    __bf16* kpre = (__bf16*)(ws + off);

    // 1) transpose+convert activations; convert weights
    transpose_cvt<<<dim3(HWP_ / 32, C_ / 32, N_), 256, 0, stream>>>(x, xt);
    int n4w = C_ * C_ / 4;
    cvt_f32_to_bf16<<<(n4w + 255) / 256, 256, 0, stream>>>(qw, wqb, n4w);
    cvt_f32_to_bf16<<<(n4w + 255) / 256, 256, 0, stream>>>(kw, wkb, n4w);
    cvt_f32_to_bf16<<<(n4w + 255) / 256, 256, 0, stream>>>(pw, wpb, n4w);

    // 2) fused Q+K WMMA GEMM + BN (bf16 out)
    dim3 gg(HWP_ / 32, C_ / 64, N_);   // (7, 6, 128)
    gemm_qk<<<gg, 256, 0, stream>>>(xt, wqb, wkb,
                                    qg, qbe, qm, qv, kg, kbe, km, kv,
                                    qpre, kpre);

    // 3) fused LIF + attention + gating, y written transposed over xt region
    lif_attn<<<B_ * HEADS_, 256, 0, stream>>>(qpre, kpre, xt);

    // 4) projection GEMM (+bias+BN) fused with final LIF -> d_out
    dim3 gp(HWP_ / 32, C_ / 64, B_);   // (7, 6, 32)
    gemm_proj_lif<<<gp, 256, 0, stream>>>(xt, wpb, pg, pbe, pm, pv, pb,
                                          (float*)d_out);
}
